// MyLSTM_79096117723356
// MI455X (gfx1250) — matrix operand order repaired and checked by reading, not compiled, not run on hardware
//
#include <hip/hip_runtime.h>
#include <hip/hip_bf16.h>
#include <cstdint>
#include <cstddef>

// Problem dims (compile-time constants from the reference)
#define B_ 64
#define S_ 512
#define E_ 1024
#define H_ 1024
#define K_ (E_ + H_)   // 2048 : concat [x_t, h_{t-1}]

typedef __bf16 bf16_t;
typedef __attribute__((ext_vector_type(16))) __bf16 v16bf;
typedef __attribute__((ext_vector_type(8)))  float  v8f;

// ---------------------------------------------------------------------------
// Fast device math: single-trans-op rcp / tanh (both exist on gfx1250).
// ---------------------------------------------------------------------------
__device__ __forceinline__ float fast_rcp(float x) {
#if __has_builtin(__builtin_amdgcn_rcpf)
    return __builtin_amdgcn_rcpf(x);          // v_rcp_f32
#else
    return 1.0f / x;
#endif
}
__device__ __forceinline__ float sigmoid_f(float x) {
    return fast_rcp(1.0f + __expf(-x));       // v_exp_f32 + v_rcp_f32
}
__device__ __forceinline__ float tanh_f(float x) {
#if __has_builtin(__builtin_amdgcn_tanhf)
    return __builtin_amdgcn_tanhf(x);         // v_tanh_f32 (confirmed in asm)
#else
    return __builtin_fmaf(2.0f, sigmoid_f(2.0f * x), -1.0f);
#endif
}

// ---------------------------------------------------------------------------
// A-fragment handling. The 16 bf16 elements a lane holds for a
// 16x32 A tile are one contiguous run of 16 K-values (pa..pa+15).
// Prefetched loads stay as raw float4s; cvt to bf16 happens at consumption.
// ---------------------------------------------------------------------------
struct ARaw { float4 r0, r1, r2, r3; };

__device__ __forceinline__ ARaw load_a_raw(const float* __restrict__ pa) {
    ARaw t;
    t.r0 = *(const float4*)(pa +  0);
    t.r1 = *(const float4*)(pa +  4);
    t.r2 = *(const float4*)(pa +  8);
    t.r3 = *(const float4*)(pa + 12);
    return t;
}

__device__ __forceinline__ v16bf cvt_a(const ARaw& t) {
    v16bf a;
    a[0]=(bf16_t)t.r0.x; a[1]=(bf16_t)t.r0.y; a[2]=(bf16_t)t.r0.z; a[3]=(bf16_t)t.r0.w;
    a[4]=(bf16_t)t.r1.x; a[5]=(bf16_t)t.r1.y; a[6]=(bf16_t)t.r1.z; a[7]=(bf16_t)t.r1.w;
    a[8]=(bf16_t)t.r2.x; a[9]=(bf16_t)t.r2.y; a[10]=(bf16_t)t.r2.z; a[11]=(bf16_t)t.r2.w;
    a[12]=(bf16_t)t.r3.x; a[13]=(bf16_t)t.r3.y; a[14]=(bf16_t)t.r3.z; a[15]=(bf16_t)t.r3.w;
    return a;
}

// 4 gate WMMAs sharing one A fragment.
__device__ __forceinline__ void wmma4(const v16bf& a,
                                      const v16bf& b0, const v16bf& b1,
                                      const v16bf& b2, const v16bf& b3,
                                      v8f& aF, v8f& aI, v8f& aC, v8f& aO) {
    aF = __builtin_amdgcn_wmma_f32_16x16x32_bf16(false, a, false, b0, (short)0, aF, false, false);
    aI = __builtin_amdgcn_wmma_f32_16x16x32_bf16(false, a, false, b1, (short)0, aI, false, false);
    aC = __builtin_amdgcn_wmma_f32_16x16x32_bf16(false, a, false, b2, (short)0, aC, false, false);
    aO = __builtin_amdgcn_wmma_f32_16x16x32_bf16(false, a, false, b3, (short)0, aO, false, false);
}

// ---------------------------------------------------------------------------
// One-time per launch: fp32 gate weights [H][K] x4 -> bf16 workspace.
// 48 MB of traffic -> ~2us at 23.3 TB/s, negligible vs 512 steps.
// ---------------------------------------------------------------------------
__global__ void cvt_weights(const float* __restrict__ wf, const float* __restrict__ wi,
                            const float* __restrict__ wc, const float* __restrict__ wo,
                            bf16_t* __restrict__ out) {
    const size_t per = (size_t)H_ * K_;
    size_t idx = (size_t)blockIdx.x * blockDim.x + threadIdx.x;   // exact grid
    const float* src = wf;
    size_t off = idx;
    if (idx >= 3 * per)      { src = wo; off = idx - 3 * per; }
    else if (idx >= 2 * per) { src = wc; off = idx - 2 * per; }
    else if (idx >= per)     { src = wi; off = idx - per; }
    out[idx] = (bf16_t)src[off];
}

// ---------------------------------------------------------------------------
// One LSTM timestep. Grid: 64 blocks x 128 threads = 256 wave32s.
// Each wave owns one 16x16 output tile (batch-rows x hidden-cols) for all 4
// gates and fuses sigmoid/tanh + c/h update in registers. h_{t-1} is read
// from ys[:, s-1, :] (previous launch on the stream); weights are bf16 and
// L2-resident (16 MB < 192 MB L2). The K-loop is double-buffered and split
// into two affine segments (x-region, h-region) so prefetch addresses are
// pure induction + immediate offsets: no per-iteration selects, letting the
// scheduler clause the loads and use partial loadcnt waits.
// ---------------------------------------------------------------------------
__launch_bounds__(128, 1)
__global__ void lstm_step(const float* __restrict__ x,        // [B, S, E]
                          const bf16_t* __restrict__ W4,      // [4][H][K] bf16
                          const float* __restrict__ bF, const float* __restrict__ bI,
                          const float* __restrict__ bC, const float* __restrict__ bO,
                          float* ys,                          // [B, S, H] (read s-1, write s)
                          float* __restrict__ c_state,        // [B, H]
                          int s) {
    const int lane = threadIdx.x & 31;
    const int wave = threadIdx.x >> 5;
    const int tile = blockIdx.x * 4 + wave;   // 0..255
    const int m0 = (tile & 3) * 16;           // batch-tile origin (4 tiles)
    const int n0 = (tile >> 2) * 16;          // hidden-tile origin (64 tiles)
    const bool hi = lane >= 16;
    const int r16 = lane & 15;

    const int n    = n0 + r16;                // this lane's output column
    const int arow = m0 + r16;                // this lane's A-matrix row (batch)

    const int kA = hi ? 16 : 0;    // A-frag K sub-offset per half-wave
    const int kB = hi ? 16 : 0;    // B-frag K sub-offset per half-wave

    // A sources: x[arow][s][:] for K<E ; ys[arow][s-1][:] (=h_{t-1}) for K>=E
    const float* xrow = x  + ((size_t)arow * S_ + s) * E_ + kA;
    const float* hrow = ys + ((size_t)arow * S_ + (s - 1)) * H_ + kA;

    // B-matrix pointers: B[k][n] = W_gate[n][k]; row-major W is K-contiguous,
    // so each lane's 16 bf16 = one aligned 32-byte load, no conversion needed.
    const bf16_t* w0 = W4 + ((size_t)0 * H_ + n) * K_ + kB;
    const bf16_t* w1 = W4 + ((size_t)1 * H_ + n) * K_ + kB;
    const bf16_t* w2 = W4 + ((size_t)2 * H_ + n) * K_ + kB;
    const bf16_t* w3 = W4 + ((size_t)3 * H_ + n) * K_ + kB;

    // Fold the per-column bias into the accumulator init: every element of a
    // lane's v8f C-fragment lives in the same output column n.
    const float bf_v = bF[n], bi_v = bI[n], bc_v = bC[n], bo_v = bO[n];
    v8f accF, accI, accC, accO;
    #pragma unroll
    for (int r = 0; r < 8; ++r) {
        accF[r] = bf_v; accI[r] = bi_v; accC[r] = bc_v; accO[r] = bo_v;
    }

    // ---- software-pipelined K loop: double-buffered, two affine segments ---
    ARaw  aR  = load_a_raw(xrow);               // slice k0 = 0
    v16bf b0c = *(const v16bf*)(w0);
    v16bf b1c = *(const v16bf*)(w1);
    v16bf b2c = *(const v16bf*)(w2);
    v16bf b3c = *(const v16bf*)(w3);

    // Segment 1: x-region. Prefetch kn = 32 .. E-32 (affine in xrow).
    for (int kn = 32; kn < E_; kn += 32) {
        ARaw  aRn = load_a_raw(xrow + kn);
        v16bf b0n = *(const v16bf*)(w0 + kn);
        v16bf b1n = *(const v16bf*)(w1 + kn);
        v16bf b2n = *(const v16bf*)(w2 + kn);
        v16bf b3n = *(const v16bf*)(w3 + kn);
        v16bf a = cvt_a(aR);
        wmma4(a, b0c, b1c, b2c, b3c, accF, accI, accC, accO);
        aR = aRn; b0c = b0n; b1c = b1n; b2c = b2n; b3c = b3n;
    }

    if (s > 0) {    // h-region contributes only when h_{t-1} != 0
        // Boundary slice: prefetch first h slice while computing last x slice.
        {
            ARaw  aRn = load_a_raw(hrow);
            v16bf b0n = *(const v16bf*)(w0 + E_);
            v16bf b1n = *(const v16bf*)(w1 + E_);
            v16bf b2n = *(const v16bf*)(w2 + E_);
            v16bf b3n = *(const v16bf*)(w3 + E_);
            v16bf a = cvt_a(aR);
            wmma4(a, b0c, b1c, b2c, b3c, accF, accI, accC, accO);
            aR = aRn; b0c = b0n; b1c = b1n; b2c = b2n; b3c = b3n;
        }
        // Segment 2: h-region. Prefetch kn = E+32 .. K-32 (affine in hrow).
        for (int kn = E_ + 32; kn < K_; kn += 32) {
            ARaw  aRn = load_a_raw(hrow + (kn - E_));
            v16bf b0n = *(const v16bf*)(w0 + kn);
            v16bf b1n = *(const v16bf*)(w1 + kn);
            v16bf b2n = *(const v16bf*)(w2 + kn);
            v16bf b3n = *(const v16bf*)(w3 + kn);
            v16bf a = cvt_a(aR);
            wmma4(a, b0c, b1c, b2c, b3c, accF, accI, accC, accO);
            aR = aRn; b0c = b0n; b1c = b1n; b2c = b2n; b3c = b3n;
        }
    }
    {   // Tail: compute last buffered slice.
        v16bf a = cvt_a(aR);
        wmma4(a, b0c, b1c, b2c, b3c, accF, accI, accC, accO);
    }

    // Epilogue: C/D layout -> element r of v8f is row m0+(hi?8:0)+r, col n.
    const int mr = m0 + (hi ? 8 : 0);
    #pragma unroll
    for (int r = 0; r < 8; ++r) {
        const int    b    = mr + r;                 // batch index
        const size_t cidx = (size_t)b * H_ + n;
        const float fv = sigmoid_f(accF[r]);
        const float iv = sigmoid_f(accI[r]);
        const float gv = tanh_f   (accC[r]);
        const float ov = sigmoid_f(accO[r]);
        const float c_old = (s > 0) ? c_state[cidx] : 0.0f;
        const float c_new = __builtin_fmaf(fv, c_old, iv * gv);
        const float h_new = ov * tanh_f(c_new);
        c_state[cidx] = c_new;
        ys[((size_t)b * S_ + s) * H_ + n] = h_new;
    }
}

// ---------------------------------------------------------------------------
// Copy hT (= ys[:, S-1, :]) and cT (workspace c_state) to the output tail.
// ---------------------------------------------------------------------------
__global__ void finalize_hc(const float* __restrict__ ys, const float* __restrict__ c_state,
                            float* __restrict__ hT, float* __restrict__ cT) {
    const int idx = blockIdx.x * blockDim.x + threadIdx.x;   // B*H exact
    const int b = idx >> 10;            // H_ == 1024
    const int j = idx & (H_ - 1);
    hT[idx] = ys[((size_t)b * S_ + (S_ - 1)) * H_ + j];
    cT[idx] = c_state[idx];
}

// ---------------------------------------------------------------------------
extern "C" void kernel_launch(void* const* d_in, const int* in_sizes, int n_in,
                              void* d_out, int out_size, void* d_ws, size_t ws_size,
                              hipStream_t stream) {
    (void)in_sizes; (void)n_in; (void)out_size; (void)ws_size;

    const float* x  = (const float*)d_in[0];
    const float* Wf = (const float*)d_in[1];
    const float* bf = (const float*)d_in[2];
    const float* Wi = (const float*)d_in[3];
    const float* bi = (const float*)d_in[4];
    const float* Wc = (const float*)d_in[5];
    const float* bc = (const float*)d_in[6];
    const float* Wo = (const float*)d_in[7];
    const float* bo = (const float*)d_in[8];

    float* ys = (float*)d_out;                       // [B,S,H]
    float* hT = ys + (size_t)B_ * S_ * H_;           // [B,H]
    float* cT = hT + (size_t)B_ * H_;                // [B,H]

    // Workspace layout: bf16 weights [4][H][K] (16 MiB) + fp32 c-state (256 KiB)
    bf16_t* W4      = (bf16_t*)d_ws;
    float*  c_state = (float*)((char*)d_ws + (size_t)4 * H_ * K_ * sizeof(bf16_t));

    // 1) Convert weights to bf16 once per launch (deterministic, cheap).
    {
        const int total = 4 * H_ * K_;               // 8,388,608 (multiple of 256)
        cvt_weights<<<total / 256, 256, 0, stream>>>(Wf, Wi, Wc, Wo, W4);
    }

    // 2) 512 dependent timesteps; stream ordering provides the h/c barrier.
    for (int s = 0; s < S_; ++s) {
        lstm_step<<<64, 128, 0, stream>>>(x, W4, bf, bi, bc, bo, ys, c_state, s);
    }

    // 3) Emit (hT, cT).
    finalize_hc<<<(B_ * H_) / 256, 256, 0, stream>>>(ys, c_state, hT, cT);
}
